// LinearLayer_39779987096206
// MI455X (gfx1250) — compile-verified, matrix operand order checked by reading, not run on hardware
//
#include <hip/hip_runtime.h>
#include <hip/hip_bf16.h>

// ---------------------------------------------------------------------------
// Sizes (hardcoded from the reference)
// ---------------------------------------------------------------------------
#define Bq   4
#define Lq   256
#define INq  768
#define HIDq 256
#define NLq  25
#define ROWS (Bq * Lq)          // 1024

typedef __attribute__((ext_vector_type(16))) _Float16 v16h;
typedef __attribute__((ext_vector_type(8)))  _Float16 v8h;
typedef __attribute__((ext_vector_type(8)))  float    v8f;
typedef __attribute__((ext_vector_type(4)))  float    v4f;

// Load a 16x32 f16 WMMA operand fragment for this lane from a row-major
// [16 x K] (A: M-major / B: N-major) array.
// Per ISA 7.12.2: lane holds row (lane&15); halfs 0..7 = K[k0+sel .. +8),
// halfs 8..15 = K[k0+16+sel .. +8), sel = (lane>=16) ? 8 : 0.
static __device__ __forceinline__ v16h load_frag(const _Float16* rowbase,
                                                 int k0, int sel) {
    v8h lo = *(const v8h*)(rowbase + k0 + sel);
    v8h hi = *(const v8h*)(rowbase + k0 + 16 + sel);
    v16h r;
#pragma unroll
    for (int e = 0; e < 8; ++e) { r[e] = lo[e]; r[8 + e] = hi[e]; }
    return r;
}

// ---------------------------------------------------------------------------
// f32 -> f16 conversion helpers (workspace staging)
// ---------------------------------------------------------------------------
__global__ void cvt_f32_to_f16(const float* __restrict__ src,
                               _Float16* __restrict__ dst, int n) {
    int idx = blockIdx.x * blockDim.x + threadIdx.x;
    if (idx < n) dst[idx] = (_Float16)src[idx];
}

// fc3_w (25 x 256) -> zero-padded (32 x 256) f16
__global__ void cvt_fc3_pad(const float* __restrict__ src,
                            _Float16* __restrict__ dst) {
    int idx = blockIdx.x * blockDim.x + threadIdx.x;   // 32*256 threads
    int row = idx >> 8;
    int col = idx & 255;
    dst[idx] = (row < NLq) ? (_Float16)src[row * HIDq + col] : (_Float16)0.0f;
}

// ---------------------------------------------------------------------------
// Stage 1: Hh[1024][512] = f16( relu( X @ [Ws;Wt]^T + [bs;bt] ) )
//   one wave per 16x16 tile; M = row, N = o (0..511), K = 768
// ---------------------------------------------------------------------------
__global__ void __launch_bounds__(256)
stage1_kernel(const _Float16* __restrict__ Xh,    // [1024][768]
              const _Float16* __restrict__ W1h,   // [512][768]  (N-major)
              const float* __restrict__ sb,       // [256]
              const float* __restrict__ tb,       // [256]
              _Float16* __restrict__ Hh) {        // [1024][512]
    const int lane = threadIdx.x & 31;
    const int wave = threadIdx.x >> 5;
    const int tile = blockIdx.x * 8 + wave;       // 0..2047
    const int mt = tile >> 5;                     // 64 row tiles
    const int nt = tile & 31;                     // 32 out tiles
    const int sel = (lane >= 16) ? 8 : 0;
    const int r = mt * 16 + (lane & 15);          // A row (M = lane&15)
    const int o = nt * 16 + (lane & 15);          // B row (N = lane&15)

    v8f acc = {};
#pragma unroll 4
    for (int k0 = 0; k0 < INq; k0 += 32) {
        v16h a = load_frag(Xh  + r * INq, k0, sel);
        v16h b = load_frag(W1h + o * INq, k0, sel);
        acc = __builtin_amdgcn_wmma_f32_16x16x32_f16(
                  false, a, false, b, (short)0, acc, false, false);
    }

    const float bias = (o < HIDq) ? sb[o] : tb[o - HIDq];
    const int rbase = mt * 16 + ((lane >= 16) ? 8 : 0);   // D: M = v + 8*(lane>=16)
#pragma unroll
    for (int v = 0; v < 8; ++v) {
        float x = acc[v] + bias;
        x = fmaxf(x, 0.0f);
        Hh[(rbase + v) * 512 + o] = (_Float16)x;
    }
}

// ---------------------------------------------------------------------------
// Stage 2: s2 = Hs @ w_s^T ; t2 = Ht @ w_t^T + b2   (both f32, [1024][256])
//   one wave per 16x16 tile; which selects s/t half; K = 256
// ---------------------------------------------------------------------------
__global__ void __launch_bounds__(256)
stage2_kernel(const _Float16* __restrict__ Hh,    // [1024][512]
              const _Float16* __restrict__ W2h,   // [256][512] = fc2_w layout
              const float* __restrict__ b2,       // [256]
              float* __restrict__ s2,             // [1024][256]
              float* __restrict__ t2) {           // [1024][256]
    const int lane = threadIdx.x & 31;
    const int wave = threadIdx.x >> 5;
    const int tile = blockIdx.x * 8 + wave;       // 0..2047
    const int which = tile >> 10;                 // 0 = s, 1 = t
    const int mt = (tile >> 4) & 63;
    const int nt = tile & 15;
    const int sel = (lane >= 16) ? 8 : 0;
    const int r = mt * 16 + (lane & 15);
    const int o = nt * 16 + (lane & 15);
    const int koff = which * HIDq;                // column offset into 512-wide rows

    v8f acc = {};
#pragma unroll
    for (int k0 = 0; k0 < HIDq; k0 += 32) {
        v16h a = load_frag(Hh  + r * 512 + koff, k0, sel);
        v16h b = load_frag(W2h + o * 512 + koff, k0, sel);
        acc = __builtin_amdgcn_wmma_f32_16x16x32_f16(
                  false, a, false, b, (short)0, acc, false, false);
    }

    float* dst = which ? t2 : s2;
    const float bias = which ? b2[o] : 0.0f;
    const int rbase = mt * 16 + ((lane >= 16) ? 8 : 0);
#pragma unroll
    for (int v = 0; v < 8; ++v)
        dst[(rbase + v) * HIDq + o] = acc[v] + bias;
}

// ---------------------------------------------------------------------------
// Stage 3 (dominant): out[b,i,n,j] = sum_h fc3[n,h] * relu(s2[b,i,h]+t2[b,j,h])
//   one block per (b,i); GEMM with M = n (2 tiles, fc3 padded to 32),
//   N = j (each wave owns 2 j-tiles of 16), K = h (8 steps of 32).
//   k-loop outermost so that per k-step:
//     - fc3 A-fragments loaded ONCE, reused by 4 WMMAs
//     - broadcast s2[i,.] values loaded ONCE, reused by both B-fragments
//     - two B-fragments generated in registers (relu(s2+t2) -> f16)
//   D lanes = j -> fully coalesced f32 stores along innermost output dim.
// ---------------------------------------------------------------------------
__global__ void __launch_bounds__(256)
stage3_kernel(const float* __restrict__ s2,       // [1024][256]
              const float* __restrict__ t2,       // [1024][256] (b-major)
              const _Float16* __restrict__ fc3h,  // [32][256] zero-padded
              float* __restrict__ out) {          // [b][i][25][256]
    const int lane = threadIdx.x & 31;
    const int wave = threadIdx.x >> 5;            // 0..7
    const int bi = blockIdx.x;                    // 0..1023 = b*256 + i
    const int b  = bi >> 8;
    const int sel = (lane >= 16) ? 8 : 0;
    const int jl  = lane & 15;

    const float* __restrict__ s2row = s2 + bi * HIDq;
    const float* __restrict__ t2b   = t2 + b * Lq * HIDq;

    const int jt0 = wave * 2;
    const int jt1 = wave * 2 + 1;
    const float* __restrict__ t2r0 = t2b + (jt0 * 16 + jl) * HIDq;
    const float* __restrict__ t2r1 = t2b + (jt1 * 16 + jl) * HIDq;
    const _Float16* __restrict__ arow0 = fc3h + jl * HIDq;         // n = jl
    const _Float16* __restrict__ arow1 = fc3h + (16 + jl) * HIDq;  // n = 16+jl

    v8f acc00 = {}, acc01 = {};   // j-tile 0: n-tiles 0,1
    v8f acc10 = {}, acc11 = {};   // j-tile 1: n-tiles 0,1

#pragma unroll
    for (int k0 = 0; k0 < HIDq; k0 += 32) {
        const int h1 = k0 + sel;
        const int h2 = k0 + 16 + sel;

        // shared across both j-tiles: s2 broadcast values + fc3 A-fragments
        v4f sa = *(const v4f*)(s2row + h1);
        v4f sb = *(const v4f*)(s2row + h1 + 4);
        v4f sc = *(const v4f*)(s2row + h2);
        v4f sd = *(const v4f*)(s2row + h2 + 4);
        v16h a0 = load_frag(arow0, k0, sel);
        v16h a1 = load_frag(arow1, k0, sel);

        // ---- j-tile 0: generate B fragment, 2 WMMAs ----
        {
            v4f ta = *(const v4f*)(t2r0 + h1);
            v4f tb = *(const v4f*)(t2r0 + h1 + 4);
            v4f tc = *(const v4f*)(t2r0 + h2);
            v4f td = *(const v4f*)(t2r0 + h2 + 4);
            v16h bf;
#pragma unroll
            for (int e = 0; e < 4; ++e) {
                bf[e]      = (_Float16)fmaxf(sa[e] + ta[e], 0.0f);
                bf[4 + e]  = (_Float16)fmaxf(sb[e] + tb[e], 0.0f);
                bf[8 + e]  = (_Float16)fmaxf(sc[e] + tc[e], 0.0f);
                bf[12 + e] = (_Float16)fmaxf(sd[e] + td[e], 0.0f);
            }
            acc00 = __builtin_amdgcn_wmma_f32_16x16x32_f16(
                        false, a0, false, bf, (short)0, acc00, false, false);
            acc01 = __builtin_amdgcn_wmma_f32_16x16x32_f16(
                        false, a1, false, bf, (short)0, acc01, false, false);
        }

        // ---- j-tile 1: generate B fragment, 2 WMMAs ----
        {
            v4f ta = *(const v4f*)(t2r1 + h1);
            v4f tb = *(const v4f*)(t2r1 + h1 + 4);
            v4f tc = *(const v4f*)(t2r1 + h2);
            v4f td = *(const v4f*)(t2r1 + h2 + 4);
            v16h bf;
#pragma unroll
            for (int e = 0; e < 4; ++e) {
                bf[e]      = (_Float16)fmaxf(sa[e] + ta[e], 0.0f);
                bf[4 + e]  = (_Float16)fmaxf(sb[e] + tb[e], 0.0f);
                bf[8 + e]  = (_Float16)fmaxf(sc[e] + tc[e], 0.0f);
                bf[12 + e] = (_Float16)fmaxf(sd[e] + td[e], 0.0f);
            }
            acc10 = __builtin_amdgcn_wmma_f32_16x16x32_f16(
                        false, a0, false, bf, (short)0, acc10, false, false);
            acc11 = __builtin_amdgcn_wmma_f32_16x16x32_f16(
                        false, a1, false, bf, (short)0, acc11, false, false);
        }
    }

    // store: D lane = N = j (coalesced over j), VGPR v -> n = v + 8*(lane>=16)
    const int nb = (lane >= 16) ? 8 : 0;
    {
        float* __restrict__ obase = out + (bi * NLq) * Lq + (jt0 * 16 + jl);
#pragma unroll
        for (int v = 0; v < 8; ++v)
            obase[(nb + v) * Lq] = acc00[v];              // n = 0..15
#pragma unroll
        for (int v = 0; v < 8; ++v) {
            const int n = 16 + nb + v;
            if (n < NLq) obase[n * Lq] = acc01[v];        // n = 16..24
        }
    }
    {
        float* __restrict__ obase = out + (bi * NLq) * Lq + (jt1 * 16 + jl);
#pragma unroll
        for (int v = 0; v < 8; ++v)
            obase[(nb + v) * Lq] = acc10[v];
#pragma unroll
        for (int v = 0; v < 8; ++v) {
            const int n = 16 + nb + v;
            if (n < NLq) obase[n * Lq] = acc11[v];
        }
    }
}

// ---------------------------------------------------------------------------
// Launch
// ---------------------------------------------------------------------------
extern "C" void kernel_launch(void* const* d_in, const int* in_sizes, int n_in,
                              void* d_out, int out_size, void* d_ws, size_t ws_size,
                              hipStream_t stream) {
    const float* x     = (const float*)d_in[0];   // (4,256,768)
    const float* s_w   = (const float*)d_in[1];   // (256,768)
    const float* s_b   = (const float*)d_in[2];   // (256)
    const float* t_w   = (const float*)d_in[3];   // (256,768)
    const float* t_b   = (const float*)d_in[4];   // (256)
    const float* fc2_w = (const float*)d_in[5];   // (256,512)
    const float* fc2_b = (const float*)d_in[6];   // (256)
    const float* fc3_w = (const float*)d_in[7];   // (25,256)
    float* out = (float*)d_out;

    char* ws = (char*)d_ws;
    _Float16* Xh   = (_Float16*)(ws);                       // 1024*768 f16
    _Float16* W1h  = (_Float16*)(ws + 1572864);             // 512*768  f16
    _Float16* W2h  = (_Float16*)(ws + 2359296);             // 256*512  f16
    _Float16* F3h  = (_Float16*)(ws + 2621440);             // 32*256   f16
    _Float16* Hh   = (_Float16*)(ws + 2637824);             // 1024*512 f16
    float*    s2   = (float*)   (ws + 3686400);             // 1024*256 f32
    float*    t2   = (float*)   (ws + 4734976);             // 1024*256 f32

    // f32 -> f16 staging
    cvt_f32_to_f16<<<(ROWS * INq + 255) / 256, 256, 0, stream>>>(x, Xh, ROWS * INq);
    cvt_f32_to_f16<<<(HIDq * INq + 255) / 256, 256, 0, stream>>>(s_w, W1h, HIDq * INq);
    cvt_f32_to_f16<<<(HIDq * INq + 255) / 256, 256, 0, stream>>>(t_w, W1h + HIDq * INq, HIDq * INq);
    cvt_f32_to_f16<<<(HIDq * 512 + 255) / 256, 256, 0, stream>>>(fc2_w, W2h, HIDq * 512);
    cvt_fc3_pad<<<(32 * HIDq) / 256, 256, 0, stream>>>(fc3_w, F3h);

    // Stage 1: 2048 tiles (64 x 32), 8 waves/block
    stage1_kernel<<<256, 256, 0, stream>>>(Xh, W1h, s_b, t_b, Hh);

    // Stage 2: 2048 tiles (2 x 64 x 16), 8 waves/block
    stage2_kernel<<<256, 256, 0, stream>>>(Hh, W2h, fc2_b, s2, t2);

    // Stage 3: one block per (b,i)
    stage3_kernel<<<ROWS, 256, 0, stream>>>(s2, t2, F3h, out);
}
